// PointMLP_41704132444910
// MI455X (gfx1250) — matrix operand order repaired and checked by reading, not compiled, not run on hardware
//
#include <hip/hip_runtime.h>
#include <math.h>

// ---------------------------------------------------------------------------
// PointMLP forward for MI455X (gfx1250).
// All 1x1-conv matmuls use V_WMMA_F32_16X16X32_F16 (f16 in / f32 accumulate).
// Weights (with BN scale folded) are converted to f16 once per call into d_ws.
// Activations are staged channels-last in LDS; each group / point-tile runs its
// whole residual chain out of LDS (<=64KB, WGP has 320KB).
// Convs use 2x2 register-blocked WMMA tiles (4 independent accumulators,
// A/B fragment reuse) when COUT/32 >= nwaves, else 1x1 tiles.
// ---------------------------------------------------------------------------

typedef __attribute__((ext_vector_type(16))) _Float16 v16h;
typedef __attribute__((ext_vector_type(8)))  _Float16 v8h;
typedef __attribute__((ext_vector_type(8)))  float    v8f;

#define BB   16
#define N0P  2048
#define C0P  32
#define KNEI 32

extern __shared__ _Float16 smem[];

// A-fragment load: halves 0..15 = K at p..p+15 (contiguous, matches B)
__device__ __forceinline__ v16h load_afrag(const _Float16* p)
{
  v8h lo  = *(const v8h*)p;
  v8h hi8 = *(const v8h*)(p + 8);
  return __builtin_shufflevector(lo, hi8, 0, 1, 2, 3, 4, 5, 6, 7,
                                          8, 9, 10, 11, 12, 13, 14, 15);
}

__device__ __forceinline__ void store_tile(const v8f& acc, _Float16* dst,
                                           const _Float16* res,
                                           const float* __restrict__ bias,
                                           bool do_relu, int COUT,
                                           int mrow, int j)
{
#pragma unroll
  for (int v = 0; v < 8; v++) {
    const int m2 = mrow + v;                         // C/D layout (ISA 7.12.2)
    float val = acc[v] + bias[m2];
    if (res)     val += (float)res[(size_t)j * COUT + m2];
    if (do_relu) val  = fmaxf(val, 0.f);
    dst[(size_t)j * COUT + m2] = (_Float16)val;
  }
}

// ---------------------------------------------------------------------------
// Shared WMMA conv: dst(32 x COUT) = act(relu?)( W(COUT x CIN2) * src(CIN2 x 32)
//                                               + bias [+ res] )
// src/dst/res are LDS, point/neighbor-major (row j holds CIN2 or COUT halves).
// res == dst is allowed (element-wise read-before-write by the owning lane).
// ---------------------------------------------------------------------------
__device__ __forceinline__ void wmma_conv(
    const _Float16* __restrict__ W, const float* __restrict__ bias,
    const _Float16* src, _Float16* dst, const _Float16* res,
    bool do_relu, int CIN2, int COUT, int lane, int wave, int nwaves)
{
  const int n    = lane & 15;
  const bool hi  = lane >= 16;
  const int klo  = hi ? 16 : 0;                 // A-matrix K sub-offset
  const int boff = hi ? 16 : 0;                 // B-matrix K sub-offset
  const int mhi  = hi ? 8 : 0;                  // C/D row sub-offset
  const int nsup = COUT >> 5;                   // 2x2 supertiles (32 rows each)

  const _Float16* srow0 = src + (size_t)n * CIN2 + boff;         // cols 0..15
  const _Float16* srow1 = src + (size_t)(16 + n) * CIN2 + boff;  // cols 16..31

  if (nsup >= nwaves) {
    // ---- 2x2 register-blocked path: 4 WMMAs per 4 fragment loads ----
    for (int st = wave; st < nsup; st += nwaves) {
      const int m0 = st << 5;
      const _Float16* wrow0 = W + (size_t)(m0 + n) * CIN2 + klo;
      const _Float16* wrow1 = W + (size_t)(m0 + 16 + n) * CIN2 + klo;
      v8f a00 = {}, a01 = {}, a10 = {}, a11 = {};
      for (int c0 = 0; c0 < CIN2; c0 += 32) {
        v16h af0 = load_afrag(wrow0 + c0);
        v16h af1 = load_afrag(wrow1 + c0);
        v16h bf0 = *(const v16h*)(srow0 + c0);
        v16h bf1 = *(const v16h*)(srow1 + c0);
        a00 = __builtin_amdgcn_wmma_f32_16x16x32_f16(false, af0, false, bf0,
                                                     (short)0, a00, false, false);
        a01 = __builtin_amdgcn_wmma_f32_16x16x32_f16(false, af0, false, bf1,
                                                     (short)0, a01, false, false);
        a10 = __builtin_amdgcn_wmma_f32_16x16x32_f16(false, af1, false, bf0,
                                                     (short)0, a10, false, false);
        a11 = __builtin_amdgcn_wmma_f32_16x16x32_f16(false, af1, false, bf1,
                                                     (short)0, a11, false, false);
      }
      store_tile(a00, dst, res, bias, do_relu, COUT, m0 + mhi,      n);
      store_tile(a01, dst, res, bias, do_relu, COUT, m0 + mhi,      16 + n);
      store_tile(a10, dst, res, bias, do_relu, COUT, m0 + 16 + mhi, n);
      store_tile(a11, dst, res, bias, do_relu, COUT, m0 + 16 + mhi, 16 + n);
    }
  } else {
    // ---- 1x1 path (small layers; keeps all waves busy) ----
    const int ntiles = (COUT >> 4) << 1;
    for (int t = wave; t < ntiles; t += nwaves) {
      const int mt = t >> 1, nt = t & 1;
      const _Float16* wrow = W + (size_t)(mt * 16 + n) * CIN2 + klo;
      const _Float16* srow = nt ? srow1 : srow0;
      v8f acc = {};
      for (int c0 = 0; c0 < CIN2; c0 += 32) {
        v16h af = load_afrag(wrow + c0);
        v16h bf = *(const v16h*)(srow + c0);
        acc = __builtin_amdgcn_wmma_f32_16x16x32_f16(false, af, false, bf,
                                                     (short)0, acc, false, false);
      }
      store_tile(acc, dst, res, bias, do_relu, COUT, mt * 16 + mhi, nt * 16 + n);
    }
  }
}

// ---------------------------------------------------------------------------
// Fold BN scale into weights, convert to f16:  dst[o,c] = (h)(W[o,c]*s[o])
// ---------------------------------------------------------------------------
__global__ void fold_w_kernel(const float* __restrict__ W,
                              const float* __restrict__ s,
                              _Float16* __restrict__ dst, int cout, int cin)
{
  int i = blockIdx.x * blockDim.x + threadIdx.x;
  if (i < cout * cin) dst[i] = (_Float16)(W[i] * s[i / cin]);
}

// ---------------------------------------------------------------------------
// Encoder: feat[p,c] = relu((W[c,:]*xyz[p,:])*s[c] + b[c]),  feat (B*N, 32)
// ---------------------------------------------------------------------------
__global__ void enc_kernel(const float* __restrict__ xyz,
                           const float* __restrict__ W,
                           const float* __restrict__ sc,
                           const float* __restrict__ bi,
                           float* __restrict__ feat, int total)
{
  int p = blockIdx.x * blockDim.x + threadIdx.x;
  if (p >= total) return;
  float x = xyz[p * 3 + 0], y = xyz[p * 3 + 1], z = xyz[p * 3 + 2];
#pragma unroll
  for (int c = 0; c < C0P; c++) {
    float v = W[c * 3 + 0] * x + W[c * 3 + 1] * y + W[c * 3 + 2] * z;
    feat[(size_t)p * C0P + c] = fmaxf(v * sc[c] + bi[c], 0.f);
  }
}

// ---------------------------------------------------------------------------
// Farthest point sampling: one workgroup per batch, points resident in LDS.
// idx[b,0]=0; then npoint-1 dependent argmax(min-dist) steps (matches scan).
// ---------------------------------------------------------------------------
__global__ void fps_kernel(const float* __restrict__ xyz, int N, int npoint,
                           int* __restrict__ idxOut)
{
  __shared__ float sx[N0P], sy[N0P], sz[N0P], minD[N0P];
  __shared__ float rv[256];
  __shared__ int   ri[256];
  __shared__ int   sLast;
  const int b = blockIdx.x, tid = threadIdx.x, nth = blockDim.x;
  const float* base = xyz + (size_t)b * N * 3;
  for (int i = tid; i < N; i += nth) {
    sx[i] = base[i * 3]; sy[i] = base[i * 3 + 1]; sz[i] = base[i * 3 + 2];
    minD[i] = 3.4e38f;
  }
  if (tid == 0) sLast = 0;
  __syncthreads();
  for (int t = 0; t < npoint; t++) {
    const int last = sLast;
    if (tid == 0) idxOut[(size_t)b * npoint + t] = last;
    const float px = sx[last], py = sy[last], pz = sz[last];
    float bestV = -1.f; int bestI = 0;
    for (int i = tid; i < N; i += nth) {
      float dx = sx[i] - px, dy = sy[i] - py, dz = sz[i] - pz;
      float m = fminf(minD[i], dx * dx + dy * dy + dz * dz);
      minD[i] = m;
      if (m > bestV) { bestV = m; bestI = i; }   // strict > : first max wins
    }
    rv[tid] = bestV; ri[tid] = bestI;
    __syncthreads();
    for (int off = nth >> 1; off > 0; off >>= 1) {
      if (tid < off) {
        float v2 = rv[tid + off]; int i2 = ri[tid + off];
        if (v2 > rv[tid] || (v2 == rv[tid] && i2 < ri[tid])) { rv[tid] = v2; ri[tid] = i2; }
      }
      __syncthreads();
    }
    if (tid == 0) sLast = ri[0];
    __syncthreads();
  }
}

// ---------------------------------------------------------------------------
// kNN (K=32): one thread per center; threshold-gated insertion top-K.
// Neighbor ORDER is irrelevant downstream (1x1 convs + max-pool over K).
// Also writes new_xyz (the FPS-selected coordinates).
// ---------------------------------------------------------------------------
__global__ void knn_kernel(const float* __restrict__ xyz,
                           const int* __restrict__ fpsIdx, int N, int S,
                           int* __restrict__ nidxOut, float* __restrict__ newXyz)
{
  int t = blockIdx.x * blockDim.x + threadIdx.x;
  if (t >= BB * S) return;
  const int b = t / S;
  const float* base = xyz + (size_t)b * N * 3;
  const int ci = fpsIdx[t];
  const float cx = base[ci * 3], cy = base[ci * 3 + 1], cz = base[ci * 3 + 2];
  newXyz[(size_t)t * 3 + 0] = cx;
  newXyz[(size_t)t * 3 + 1] = cy;
  newXyz[(size_t)t * 3 + 2] = cz;
  const float cc = cx * cx + cy * cy + cz * cz;
  float kd[KNEI]; int ki[KNEI];
#pragma unroll
  for (int q = 0; q < KNEI; q++) { kd[q] = 3.4e38f; ki[q] = 0; }
  float worst = kd[KNEI - 1];
  for (int i = 0; i < N; i++) {
    float x = base[i * 3], y = base[i * 3 + 1], z = base[i * 3 + 2];
    float d = cc - 2.f * (cx * x + cy * y + cz * z) + (x * x + y * y + z * z);
    if (d < worst) {
      int p = KNEI - 1;
      while (p > 0 && kd[p - 1] > d) { kd[p] = kd[p - 1]; ki[p] = ki[p - 1]; --p; }
      kd[p] = d; ki[p] = i;
      worst = kd[KNEI - 1];
    }
  }
  for (int q = 0; q < KNEI; q++) nidxOut[(size_t)t * KNEI + q] = ki[q];
}

// ---------------------------------------------------------------------------
// Group stage: gather [neighbor|center] feats -> transfer conv -> 2 residual
// blocks -> max over K. One workgroup (8 waves) per (b,s) group.
// LDS: buf g (32 x 2cin == 32 x cout) + buf x (32 x cout), ping-pong.
// ---------------------------------------------------------------------------
__global__ __launch_bounds__(256)
void group_kernel(const float* __restrict__ feat, const int* __restrict__ nidx,
                  const int* __restrict__ fpsIdx,
                  const _Float16* Wt,  const float* bt,
                  const _Float16* Wa0, const float* ba0,
                  const _Float16* Wb0, const float* bb0,
                  const _Float16* Wa1, const float* ba1,
                  const _Float16* Wb1, const float* bb1,
                  float* __restrict__ outFeat, int Npts, int S, int CIN, int COUT)
{
  const int CIN2 = 2 * CIN;                // == COUT
  _Float16* g = smem;                      // 32 * CIN2 halves
  _Float16* x = smem + 32 * CIN2;          // 32 * COUT halves
  const int bs = blockIdx.x;
  const int b  = bs / S;
  const int tid = threadIdx.x;
  __shared__ int nb[KNEI];
  if (tid < KNEI) nb[tid] = nidx[(size_t)bs * KNEI + tid];
  const int center = fpsIdx[bs];
  __syncthreads();
  const float* fb = feat + (size_t)b * Npts * CIN;
  for (int e = tid; e < KNEI * CIN; e += blockDim.x) {
    int j = e / CIN, c = e - j * CIN;
    g[(size_t)j * CIN2 + c]       = (_Float16)fb[(size_t)nb[j]  * CIN + c];
    g[(size_t)j * CIN2 + CIN + c] = (_Float16)fb[(size_t)center * CIN + c];
  }
  __syncthreads();
  const int lane = tid & 31, wave = tid >> 5, nw = blockDim.x >> 5;
  wmma_conv(Wt,  bt,  g, x, nullptr, true, CIN2, COUT, lane, wave, nw); __syncthreads();
  wmma_conv(Wa0, ba0, x, g, nullptr, true, COUT, COUT, lane, wave, nw); __syncthreads();
  wmma_conv(Wb0, bb0, g, x, x,       true, COUT, COUT, lane, wave, nw); __syncthreads();
  wmma_conv(Wa1, ba1, x, g, nullptr, true, COUT, COUT, lane, wave, nw); __syncthreads();
  wmma_conv(Wb1, bb1, g, x, x,       true, COUT, COUT, lane, wave, nw); __syncthreads();
  for (int c = tid; c < COUT; c += blockDim.x) {
    float m = -3.4e38f;
    for (int j = 0; j < KNEI; j++) m = fmaxf(m, (float)x[(size_t)j * COUT + c]);
    outFeat[(size_t)bs * COUT + c] = m;
  }
}

// ---------------------------------------------------------------------------
// Pos stage: 2 residual blocks over points; one workgroup per 32 points.
// ---------------------------------------------------------------------------
__global__ __launch_bounds__(256)
void pos_kernel(const float* __restrict__ featIn,
                const _Float16* Wa0, const float* ba0,
                const _Float16* Wb0, const float* bb0,
                const _Float16* Wa1, const float* ba1,
                const _Float16* Wb1, const float* bb1,
                float* __restrict__ featOut, int COUT)
{
  _Float16* h = smem;
  _Float16* x = smem + 32 * COUT;
  const size_t p0 = (size_t)blockIdx.x * 32;
  const int tid = threadIdx.x;
  for (int e = tid; e < 32 * COUT; e += blockDim.x) {
    int j = e / COUT, c = e - j * COUT;
    x[(size_t)j * COUT + c] = (_Float16)featIn[(p0 + j) * COUT + c];
  }
  __syncthreads();
  const int lane = tid & 31, wave = tid >> 5, nw = blockDim.x >> 5;
  wmma_conv(Wa0, ba0, x, h, nullptr, true, COUT, COUT, lane, wave, nw); __syncthreads();
  wmma_conv(Wb0, bb0, h, x, x,       true, COUT, COUT, lane, wave, nw); __syncthreads();
  wmma_conv(Wa1, ba1, x, h, nullptr, true, COUT, COUT, lane, wave, nw); __syncthreads();
  wmma_conv(Wb1, bb1, h, x, x,       true, COUT, COUT, lane, wave, nw); __syncthreads();
  for (int e = tid; e < 32 * COUT; e += blockDim.x) {
    int j = e / COUT, c = e - j * COUT;
    featOut[(p0 + j) * COUT + c] = (float)x[(size_t)j * COUT + c];
  }
}

// ---------------------------------------------------------------------------
// Head MLPs (512->16->16->{2,4}); one thread per point (tiny).
// ---------------------------------------------------------------------------
__global__ void head_kernel(const float* __restrict__ f,
    const float* cW0, const float* cb0, const float* cW1, const float* cb1,
    const float* cW2, const float* cb2,
    const float* gW0, const float* gb0, const float* gW1, const float* gb1,
    const float* gW2, const float* gb2,
    float* __restrict__ outLogits, float* __restrict__ outBbox, int total)
{
  int p = blockIdx.x * blockDim.x + threadIdx.x;
  if (p >= total) return;
  const float* row = f + (size_t)p * 512;
  float h0[16], h1[16];
  // class MLP
#pragma unroll
  for (int o = 0; o < 16; o++) {
    float a = cb0[o];
    for (int i = 0; i < 512; i++) a += cW0[o * 512 + i] * row[i];
    h0[o] = fmaxf(a, 0.f);
  }
#pragma unroll
  for (int o = 0; o < 16; o++) {
    float a = cb1[o];
#pragma unroll
    for (int i = 0; i < 16; i++) a += cW1[o * 16 + i] * h0[i];
    h1[o] = fmaxf(a, 0.f);
  }
#pragma unroll
  for (int o = 0; o < 2; o++) {
    float a = cb2[o];
#pragma unroll
    for (int i = 0; i < 16; i++) a += cW2[o * 16 + i] * h1[i];
    outLogits[(size_t)p * 2 + o] = a;
  }
  // bbox MLP
#pragma unroll
  for (int o = 0; o < 16; o++) {
    float a = gb0[o];
    for (int i = 0; i < 512; i++) a += gW0[o * 512 + i] * row[i];
    h0[o] = fmaxf(a, 0.f);
  }
#pragma unroll
  for (int o = 0; o < 16; o++) {
    float a = gb1[o];
#pragma unroll
    for (int i = 0; i < 16; i++) a += gW1[o * 16 + i] * h0[i];
    h1[o] = fmaxf(a, 0.f);
  }
#pragma unroll
  for (int o = 0; o < 4; o++) {
    float a = gb2[o];
#pragma unroll
    for (int i = 0; i < 16; i++) a += gW2[o * 16 + i] * h1[i];
    outBbox[(size_t)p * 4 + o] = 1.f / (1.f + expf(-a));
  }
}

// ---------------------------------------------------------------------------
extern "C" void kernel_launch(void* const* d_in, const int* in_sizes, int n_in,
                              void* d_out, int out_size, void* d_ws, size_t ws_size,
                              hipStream_t stream)
{
  (void)in_sizes; (void)n_in; (void)out_size; (void)ws_size;
  const float* xyzIn = (const float*)d_in[0];
  auto F = [&](int i) { return (const float*)d_in[i]; };

  // ---- workspace carve-up (256B aligned) ----
  char* ws = (char*)d_ws;
  size_t off = 0;
  auto take = [&](size_t bytes) {
    size_t o = off; off += (bytes + 255) & ~(size_t)255; return o;
  };
  // f16 weights: per layer 9 matrices of cout x cout
  static const size_t layerHalfOff[4] = {0, 36864, 184320, 774144};
  _Float16* Wh     = (_Float16*)(ws + take(3133440ull * 2));
  float*    featA  = (float*)(ws + take(4194304));
  float*    featB  = (float*)(ws + take(4194304));
  float*    featM  = (float*)(ws + take(4194304));
  float*    xyzP0  = (float*)(ws + take(196608));
  float*    xyzP1  = (float*)(ws + take(196608));
  int*      fpsIdx = (int*)  (ws + take(65536));
  int*      nidx   = (int*)  (ws + take(2097152));

  float* outLogits = (float*)d_out;                    // (B,128,2)
  float* outBbox   = (float*)d_out + BB * 128 * 2;     // (B,128,4)
  float* fOut      = (float*)d_out + BB * 128 * 6;     // (B,128,512)

  // ---- fold all layer weights (BN scale folded, f32 -> f16) ----
  static const int widx[9] = {0, 3, 6, 9, 12, 15, 18, 21, 24};
  for (int l = 0; l < 4; l++) {
    const int cout = 64 << l;
    const int base = 4 + 27 * l;
    const size_t cc = (size_t)cout * cout;
    _Float16* Wl = Wh + layerHalfOff[l];
    for (int k = 0; k < 9; k++) {
      fold_w_kernel<<<(int)((cc + 255) / 256), 256, 0, stream>>>(
          F(base + widx[k]), F(base + widx[k] + 1), Wl + (size_t)k * cc, cout, cout);
    }
  }

  // ---- encoder ----
  enc_kernel<<<(BB * N0P + 255) / 256, 256, 0, stream>>>(
      xyzIn, F(1), F(2), F(3), featA, BB * N0P);

  // ---- 4 extraction layers ----
  const float* xyzCur = xyzIn;
  float* xyzNextBuf[2] = {xyzP0, xyzP1};
  float* featCur = featA;
  int N = N0P, cin = C0P;
  for (int l = 0; l < 4; l++) {
    const int cout = 2 * cin;
    const int S = N / 2;
    const int base = 4 + 27 * l;
    const size_t cc = (size_t)cout * cout;
    _Float16* Wl = Wh + layerHalfOff[l];
    float* xyzNext = xyzNextBuf[l & 1];
    float* featNext = (l == 3) ? fOut : ((featCur == featA) ? featB : featA);
    const size_t ldsBytes = (size_t)128 * cout;   // 2 buffers of 32*cout halves

    fps_kernel<<<BB, 256, 0, stream>>>(xyzCur, N, S, fpsIdx);
    knn_kernel<<<(BB * S + 255) / 256, 256, 0, stream>>>(
        xyzCur, fpsIdx, N, S, nidx, xyzNext);
    group_kernel<<<BB * S, 256, ldsBytes, stream>>>(
        featCur, nidx, fpsIdx,
        Wl + 0 * cc, F(base + 2),
        Wl + 1 * cc, F(base + 5),  Wl + 2 * cc, F(base + 8),
        Wl + 3 * cc, F(base + 11), Wl + 4 * cc, F(base + 14),
        featM, N, S, cin, cout);
    pos_kernel<<<BB * S / 32, 256, ldsBytes, stream>>>(
        featM,
        Wl + 5 * cc, F(base + 17), Wl + 6 * cc, F(base + 20),
        Wl + 7 * cc, F(base + 23), Wl + 8 * cc, F(base + 26),
        featNext, cout);

    featCur = featNext;
    xyzCur = xyzNext;
    N = S;
    cin = cout;
  }

  // ---- heads (class_mlp: d_in[112..117], bbox_mlp: d_in[118..123]) ----
  head_kernel<<<(BB * 128 + 255) / 256, 256, 0, stream>>>(
      fOut,
      F(112), F(113), F(114), F(115), F(116), F(117),
      F(118), F(119), F(120), F(121), F(122), F(123),
      outLogits, outBbox, BB * 128);
}
